// HybridHeatmapUNet_4148938408611
// MI455X (gfx1250) — matrix operand order repaired and checked by reading, not compiled, not run on hardware
//
#include <hip/hip_runtime.h>
#include <hip/hip_bf16.h>
#include <math.h>

// ---------------------------------------------------------------------------
// MI455X (gfx1250) implementation. wave32; 256-thread blocks = 8 waves.
// All channel contractions (1x1 convs, im2col 3x3 convs, transformer GEMMs,
// transposed-conv GEMMs) run through k_gemm_wmma -> v_wmma_f32_16x16x32_f16.
// All f16 operands are K-padded to a multiple of 32 so the WMMA inner loop is
// branch-free and fully vectorized (no per-lane EXEC divergence).
// ---------------------------------------------------------------------------

typedef _Float16 half_t;
typedef __attribute__((ext_vector_type(16))) _Float16 v16h;
typedef __attribute__((ext_vector_type(8)))  _Float16 v8h;
typedef __attribute__((ext_vector_type(8)))  float    v8f;

static __device__ __forceinline__ int imin(int a, int b) { return a < b ? a : b; }
static __device__ __forceinline__ int imax(int a, int b) { return a > b ? a : b; }

#define GTID ((long long)blockIdx.x * blockDim.x + threadIdx.x)

// --------------------------- operand packing (K-padded f16) ----------------

__global__ void k_cast_pad(const float* __restrict__ s, half_t* __restrict__ d,
                           long long rows, int K, int Kp) {
  long long n = rows * Kp, i = GTID; if (i >= n) return;
  int k = i % Kp; long long r = i / Kp;
  d[i] = (k < K) ? (half_t)s[r * K + k] : (half_t)0.0f;
}

__global__ void k_im2col3(const float* __restrict__ in, half_t* __restrict__ out,
                          int B, int H, int W, int C, int Kp) {
  long long n = (long long)B * H * W * Kp, i = GTID; if (i >= n) return;
  int k = i % Kp; long long pix = i / Kp;
  half_t hv = (half_t)0.0f;
  if (k < 9 * C) {
    int kk = k / C, c = k % C, ky = kk / 3, kx = kk % 3;
    int x = pix % W; long long t = pix / W; int y = t % H; int b = (int)(t / H);
    int iy = y + ky - 1, ix = x + kx - 1;
    if (iy >= 0 && iy < H && ix >= 0 && ix < W)
      hv = (half_t)in[(((long long)b * H + iy) * W + ix) * C + c];
  }
  out[i] = hv;
}

__global__ void k_pack_w3(const float* __restrict__ w, half_t* __restrict__ out,
                          int Co, int Ci, int Kp) {
  long long n = (long long)Co * Kp, i = GTID; if (i >= n) return;
  int k = i % Kp; int o = (int)(i / Kp);
  half_t v = (half_t)0.0f;
  if (k < 9 * Ci) {
    int kk = k / Ci, ci = k % Ci, ky = kk / 3, kx = kk % 3;
    v = (half_t)w[(((long long)o * Ci + ci) * 3 + ky) * 3 + kx];
  }
  out[i] = v;
}

__global__ void k_pack_wt2(const float* __restrict__ w, half_t* __restrict__ out,
                           int Ci, int Co, int Kp) {
  long long n = 4LL * Co * Kp, i = GTID; if (i >= n) return;
  int k = i % Kp; long long row = i / Kp;
  half_t v = (half_t)0.0f;
  if (k < Ci) {
    int o = row % Co; int dd = (int)(row / Co); int dy = dd >> 1, dx = dd & 1;
    v = (half_t)w[(((long long)k * Co + o) * 2 + dy) * 2 + dx];
  }
  out[i] = v;
}

// --------------------------- elementwise helpers ---------------------------

__global__ void k_nchw_to_nhwc(const float* __restrict__ in, float* __restrict__ out,
                               int B, int C, int H, int W) {
  long long n = (long long)B * C * H * W, i = GTID; if (i >= n) return;
  int x = i % W; long long t = i / W; int y = t % H; t /= H; int c = t % C; int b = t / C;
  out[(((long long)b * H + y) * W + x) * C + c] = in[i];
}

__global__ void k_nhwc_to_nchw(const float* __restrict__ in, float* __restrict__ out,
                               int B, int C, int H, int W) {
  long long n = (long long)B * C * H * W, i = GTID; if (i >= n) return;
  int c = i % C; long long t = i / C; int x = t % W; t /= W; int y = t % H; int b = t / H;
  out[(((long long)b * C + c) * H + y) * W + x] = in[i];
}

__global__ void k_add_inplace(float* __restrict__ d, const float* __restrict__ s, long long n) {
  long long i = GTID; if (i < n) d[i] += s[i];
}

// --------------------------- WMMA GEMM core --------------------------------
// C[M,N] = act(A[M,Kp] * Wt[N,Kp]^T + bias). A/Wt f16 row-major with K padded
// to Kp (multiple of 32, pad zero-filled). C f32, row stride ldc.
// act: 0=none 1=relu 2=silu. Block = 256 threads = 8 waves, tiling 64(M)x64(N);
// each wave computes 16x32 with one A fragment shared by two WMMA issues.
__global__ __launch_bounds__(256)
void k_gemm_wmma(const half_t* __restrict__ A, const half_t* __restrict__ Wt,
                 const float* __restrict__ bias, float* __restrict__ C,
                 int M, int N, int Kp, int ldc, int act) {
  const int lane = threadIdx.x & 31;
  const int wid  = threadIdx.x >> 5;
  const int wm = wid >> 1, wn = wid & 1;          // 4 x 2 wave grid
  const int m0 = blockIdx.x * 64 + wm * 16;
  const int n0 = blockIdx.y * 64 + wn * 32;
  if (m0 >= M || n0 >= N) return;                 // wave-uniform exit

  const int g = lane >> 4;                        // half-wave group
  const int r = lane & 15;
  // Clamp out-of-range rows/cols to valid ones (duplicate compute, masked store)
  const half_t* __restrict__ Ap = A  + (long long)imin(m0 + r, M - 1) * Kp;
  const half_t* __restrict__ B0 = Wt + (long long)imin(n0 + r, N - 1) * Kp;
  const half_t* __restrict__ B1 = Wt + (long long)imin(n0 + 16 + r, N - 1) * Kp;

  v8f acc0, acc1;
  #pragma unroll
  for (int i = 0; i < 8; ++i) { acc0[i] = 0.0f; acc1[i] = 0.0f; }

  for (int kb = 0; kb < Kp; kb += 32) {
    // A 16-bit 16x32 fragment: lane holds row m (=lane%16),
    // K = kb+16g+[0..15]
    v16h a = *(const v16h*)(Ap + kb + 16 * g);
    // B 16-bit 32x16 fragment: lane holds col n (=lane%16), K = kb+16g+[0..15]
    v16h b0 = *(const v16h*)(B0 + kb + 16 * g);
    v16h b1 = *(const v16h*)(B1 + kb + 16 * g);
    __builtin_prefetch(Ap + kb + 256, 0, 1);
    acc0 = __builtin_amdgcn_wmma_f32_16x16x32_f16(false, a, false, b0,
                                                  (short)0, acc0, false, false);
    acc1 = __builtin_amdgcn_wmma_f32_16x16x32_f16(false, a, false, b1,
                                                  (short)0, acc1, false, false);
  }

  // C/D layout: lane -> col (lane&15); VGPR i -> row m0+i+8*(lane>>4)
  const int col0 = n0 + r, col1 = n0 + 16 + r;
  const float bv0 = (bias && col0 < N) ? bias[col0] : 0.0f;
  const float bv1 = (bias && col1 < N) ? bias[col1] : 0.0f;
  #pragma unroll
  for (int i = 0; i < 8; ++i) {
    int row = m0 + i + 8 * g;
    if (row >= M) continue;
    float v0 = acc0[i] + bv0, v1 = acc1[i] + bv1;
    if (act == 1)      { v0 = fmaxf(v0, 0.0f); v1 = fmaxf(v1, 0.0f); }
    else if (act == 2) { v0 = v0 / (1.0f + expf(-v0)); v1 = v1 / (1.0f + expf(-v1)); }
    if (col0 < N) C[(long long)row * ldc + col0] = v0;
    if (col1 < N) C[(long long)row * ldc + col1] = v1;
  }
}

// --------------------------- conv / norm kernels (NHWC) --------------------

__global__ void k_dwconv3x3(const float* __restrict__ in, const float* __restrict__ w,
                            float* __restrict__ out, int B, int H, int W, int C) {
  long long n = (long long)B * H * W * C, i = GTID; if (i >= n) return;
  int c = i % C; long long t = i / C; int x = t % W; t /= W; int y = t % H; int b = t / H;
  float s = 0.0f;
  #pragma unroll
  for (int ky = 0; ky < 3; ++ky) {
    int iy = y + ky - 1; if (iy < 0 || iy >= H) continue;
    #pragma unroll
    for (int kx = 0; kx < 3; ++kx) {
      int ix = x + kx - 1; if (ix < 0 || ix >= W) continue;
      s += in[(((long long)b * H + iy) * W + ix) * C + c] * w[c * 9 + ky * 3 + kx];
    }
  }
  out[i] = s;
}

__global__ void k_gn_stats(const float* __restrict__ x, float* __restrict__ mv,
                           int HW, int C, int cg) {
  int bg = blockIdx.x; int b = bg >> 3; int g = bg & 7;
  const float* base = x + (long long)b * HW * C + g * cg;
  int n = HW * cg;
  float s = 0.0f, s2 = 0.0f;
  for (int i = threadIdx.x; i < n; i += 256) {
    int pix = i / cg, c = i - pix * cg;
    float v = base[(long long)pix * C + c];
    s += v; s2 += v * v;
  }
  __shared__ float r1[256], r2[256];
  r1[threadIdx.x] = s; r2[threadIdx.x] = s2; __syncthreads();
  for (int o = 128; o > 0; o >>= 1) {
    if (threadIdx.x < o) { r1[threadIdx.x] += r1[threadIdx.x + o]; r2[threadIdx.x] += r2[threadIdx.x + o]; }
    __syncthreads();
  }
  if (threadIdx.x == 0) {
    float m = r1[0] / n;
    mv[bg * 2] = m; mv[bg * 2 + 1] = r2[0] / n - m * m;
  }
}

__global__ void k_gn_apply_relu(const float* __restrict__ x, const float* __restrict__ mv,
                                const float* __restrict__ gw, const float* __restrict__ gb,
                                float* __restrict__ out, long long n, int HW, int C, int cg) {
  long long i = GTID; if (i >= n) return;
  int c = i % C; long long bp = i / C; int b = (int)(bp / HW);
  int g = c / cg;
  float m = mv[(b * 8 + g) * 2], v = mv[(b * 8 + g) * 2 + 1];
  float y = (x[i] - m) * rsqrtf(v + 1e-5f) * gw[c] + gb[c];
  out[i] = fmaxf(y, 0.0f);
}

__global__ void k_chan_mean(const float* __restrict__ x, float* __restrict__ mean,
                            int HW, int C) {
  int bc = blockIdx.x; int b = bc / C; int c = bc % C;
  const float* base = x + (long long)b * HW * C + c;
  float s = 0.0f;
  for (int i = threadIdx.x; i < HW; i += 256) s += base[(long long)i * C];
  __shared__ float r1[256];
  r1[threadIdx.x] = s; __syncthreads();
  for (int o = 128; o > 0; o >>= 1) {
    if (threadIdx.x < o) r1[threadIdx.x] += r1[threadIdx.x + o];
    __syncthreads();
  }
  if (threadIdx.x == 0) mean[bc] = r1[0] / HW;
}

__global__ void k_eca_scale(const float* __restrict__ mean, const float* __restrict__ w,
                            float* __restrict__ scale, int B, int C, int kk) {
  long long i = GTID; if (i >= (long long)B * C) return;
  int b = (int)(i / C), c = (int)(i % C);
  int half = (kk - 1) / 2;
  float s = 0.0f;
  for (int j = 0; j < kk; ++j) {
    int cc = c + j - half;
    if (cc >= 0 && cc < C) s += w[j] * mean[(long long)b * C + cc];
  }
  scale[i] = 1.0f / (1.0f + expf(-s));
}

__global__ void k_scale_ch(float* __restrict__ x, const float* __restrict__ sc,
                           long long n, int HW, int C) {
  long long i = GTID; if (i >= n) return;
  int c = i % C; int b = (int)(i / ((long long)HW * C));
  x[i] *= sc[(long long)b * C + c];
}

__global__ void k_pool2_nhwc(const float* __restrict__ in, float* __restrict__ out,
                             int B, int H, int W, int C) {
  int H2 = H >> 1, W2 = W >> 1;
  long long n = (long long)B * H2 * W2 * C, i = GTID; if (i >= n) return;
  int c = i % C; long long t = i / C; int x2 = t % W2; t /= W2; int y2 = t % H2; int b = t / H2;
  const float* p = in + (((long long)b * H + 2 * y2) * W + 2 * x2) * C + c;
  float v = fmaxf(fmaxf(p[0], p[C]), fmaxf(p[(long long)W * C], p[(long long)W * C + C]));
  out[i] = v;
}

__global__ void k_convt_scatter(const float* __restrict__ tmp, const float* __restrict__ bias,
                                float* __restrict__ out, int B, int h, int w, int Co,
                                int coff, int ctot) {
  long long n = (long long)B * h * w * 4 * Co, i = GTID; if (i >= n) return;
  int j = i % (4 * Co); long long pix = i / (4 * Co);
  int o = j % Co; int dd = j / Co; int dy = dd >> 1, dx = dd & 1;
  int x = pix % w; long long t = pix / w; int y = t % h; int b = (int)(t / h);
  out[((((long long)b * 2 * h + 2 * y + dy) * (2 * w)) + 2 * x + dx) * ctot + coff + o]
      = tmp[i] + bias[o];
}

__global__ void k_copy_ch(const float* __restrict__ src, float* __restrict__ dst,
                          long long npix, int Cs, int coff, int ctot) {
  long long n = npix * Cs, i = GTID; if (i >= n) return;
  int c = i % Cs; long long pix = i / Cs;
  dst[pix * ctot + coff + c] = src[i];
}

// --------------------------- transformer kernels ---------------------------

__global__ void k_layernorm(const float* __restrict__ in, const float* __restrict__ g,
                            const float* __restrict__ b, float* __restrict__ out, int D) {
  long long row = blockIdx.x;
  const float* p = in + row * D;
  float s = 0.0f, s2 = 0.0f;
  for (int d = threadIdx.x; d < D; d += 64) { float v = p[d]; s += v; s2 += v * v; }
  __shared__ float r1[64], r2[64];
  r1[threadIdx.x] = s; r2[threadIdx.x] = s2; __syncthreads();
  for (int o = 32; o > 0; o >>= 1) {
    if (threadIdx.x < o) { r1[threadIdx.x] += r1[threadIdx.x + o]; r2[threadIdx.x] += r2[threadIdx.x + o]; }
    __syncthreads();
  }
  float m = r1[0] / D, var = r2[0] / D - (r1[0] / D) * (r1[0] / D);
  float rstd = rsqrtf(var + 1e-5f);
  for (int d = threadIdx.x; d < D; d += 64)
    out[row * D + d] = (p[d] - m) * rstd * g[d] + b[d];
}

// grid (BP=64, heads=4, N=64), block 64 threads. qkv [R,432], o [R,144].
__global__ void k_attn(const float* __restrict__ qkv, float* __restrict__ o) {
  const int N = 64, D = 144, DH = 36, LD = 432;
  int bp = blockIdx.x, h = blockIdx.y, nq = blockIdx.z, tid = threadIdx.x;
  __shared__ float sc[64], red[64];
  const float* q  = qkv + ((long long)bp * N + nq) * LD + h * DH;
  const float* kb = qkv + (long long)bp * N * LD + D + h * DH;
  const float* vb = qkv + (long long)bp * N * LD + 2 * D + h * DH;
  float s = 0.0f;
  const float* krow = kb + (long long)tid * LD;
  for (int d = 0; d < DH; ++d) s += q[d] * krow[d];
  sc[tid] = s * (1.0f / 6.0f);     // 1/sqrt(36)
  __syncthreads();
  red[tid] = sc[tid]; __syncthreads();
  for (int off = 32; off > 0; off >>= 1) {
    if (tid < off) red[tid] = fmaxf(red[tid], red[tid + off]);
    __syncthreads();
  }
  float mx = red[0]; __syncthreads();
  float e = expf(sc[tid] - mx);
  red[tid] = e; __syncthreads();
  for (int off = 32; off > 0; off >>= 1) {
    if (tid < off) red[tid] += red[tid + off];
    __syncthreads();
  }
  float inv = 1.0f / red[0]; __syncthreads();
  sc[tid] = e * inv;
  __syncthreads();
  if (tid < DH) {
    float acc = 0.0f;
    for (int m = 0; m < N; ++m) acc += sc[m] * vb[(long long)m * LD + tid];
    o[((long long)bp * N + nq) * D + h * DH + tid] = acc;
  }
}

__global__ void k_unfold(const float* __restrict__ y, float* __restrict__ t, int C) {
  long long n = 16LL * 256 * C, i = GTID; if (i >= n) return;
  int c = i % C; long long r = i / C; int nn = r % 64; long long r2 = r / 64;
  int p = r2 % 4; int b = (int)(r2 / 4);
  int hy = nn >> 3, wx = nn & 7, dy = p >> 1, dx = p & 1;
  t[i] = y[(((long long)b * 16 + 2 * hy + dy) * 16 + 2 * wx + dx) * C + c];
}

__global__ void k_fold(const float* __restrict__ t, float* __restrict__ y, int C) {
  long long n = 16LL * 256 * C, i = GTID; if (i >= n) return;
  int c = i % C; long long r = i / C; int nn = r % 64; long long r2 = r / 64;
  int p = r2 % 4; int b = (int)(r2 / 4);
  int hy = nn >> 3, wx = nn & 7, dy = p >> 1, dx = p & 1;
  y[(((long long)b * 16 + 2 * hy + dy) * 16 + 2 * wx + dx) * C + c] = t[i];
}

// --------------------------- refine (NCHW) kernels -------------------------

__global__ void k_argmax(const float* __restrict__ coarse, float* __restrict__ maxv,
                         int* __restrict__ maxi, int HW) {
  int bk = blockIdx.x;
  const float* p = coarse + (long long)bk * HW;
  float bv = -3.4e38f; int bi = 0;
  for (int i = threadIdx.x; i < HW; i += 256) {
    float v = p[i];
    if (v > bv) { bv = v; bi = i; }
  }
  __shared__ float sv[256]; __shared__ int si[256];
  sv[threadIdx.x] = bv; si[threadIdx.x] = bi; __syncthreads();
  for (int o = 128; o > 0; o >>= 1) {
    if (threadIdx.x < o) {
      float v2 = sv[threadIdx.x + o]; int i2 = si[threadIdx.x + o];
      if (v2 > sv[threadIdx.x] || (v2 == sv[threadIdx.x] && i2 < si[threadIdx.x])) {
        sv[threadIdx.x] = v2; si[threadIdx.x] = i2;
      }
    }
    __syncthreads();
  }
  if (threadIdx.x == 0) { maxv[bk] = sv[0]; maxi[bk] = si[0]; }
}

__global__ void k_crop(const float* __restrict__ coarse, const int* __restrict__ maxi,
                       float* __restrict__ crop, int H, int W, long long n) {
  long long i = GTID; if (i >= n) return;
  int ox = i % 32; long long t = i / 32; int oy = t % 32; int bk = (int)(t / 32);
  int idx = maxi[bk]; int cy = idx / W, cx = idx % W;
  int top = imax(cy - 16, 0), bottom = imin(cy + 16, H);
  int left = imax(cx - 16, 0), right = imin(cx + 16, W);
  float Ly = (float)(bottom - top), Lx = (float)(right - left);
  float sy = fminf(fmaxf((oy + 0.5f) * Ly / 32.0f - 0.5f, 0.0f), Ly - 1.0f);
  float sx = fminf(fmaxf((ox + 0.5f) * Lx / 32.0f - 0.5f, 0.0f), Lx - 1.0f);
  int y0 = (int)floorf(sy), x0 = (int)floorf(sx);
  float wy = sy - y0, wx = sx - x0;
  int y1 = imin(y0 + 1, bottom - top - 1), x1 = imin(x0 + 1, right - left - 1);
  const float* p = coarse + (long long)bk * H * W;
  int r0 = top + y0, r1 = top + y1, c0 = left + x0, c1 = left + x1;
  float v = (1 - wy) * ((1 - wx) * p[(long long)r0 * W + c0] + wx * p[(long long)r0 * W + c1])
          +      wy  * ((1 - wx) * p[(long long)r1 * W + c0] + wx * p[(long long)r1 * W + c1]);
  crop[i] = v;
}

__global__ void k_conv_nchw(const float* __restrict__ in, const float* __restrict__ w,
                            const float* __restrict__ bias, float* __restrict__ out,
                            int Nn, int Ci, int H, int W, int Co, int ks, int pad, int relu) {
  long long n = (long long)Nn * Co * H * W, i = GTID; if (i >= n) return;
  int x = i % W; long long t = i / W; int y = t % H; t /= H; int o = t % Co; int nb = (int)(t / Co);
  float s = bias ? bias[o] : 0.0f;
  for (int ci = 0; ci < Ci; ++ci)
    for (int ky = 0; ky < ks; ++ky) {
      int iy = y + ky - pad; if (iy < 0 || iy >= H) continue;
      for (int kx = 0; kx < ks; ++kx) {
        int ix = x + kx - pad; if (ix < 0 || ix >= W) continue;
        s += in[(((long long)nb * Ci + ci) * H + iy) * W + ix]
           * w[(((long long)o * Ci + ci) * ks + ky) * ks + kx];
      }
    }
  out[i] = relu ? fmaxf(s, 0.0f) : s;
}

__global__ void k_pool2_nchw(const float* __restrict__ in, float* __restrict__ out,
                             int Nn, int C, int H, int W) {
  int H2 = H >> 1, W2 = W >> 1;
  long long n = (long long)Nn * C * H2 * W2, i = GTID; if (i >= n) return;
  int x2 = i % W2; long long t = i / W2; int y2 = t % H2; long long nc = t / H2;
  const float* p = in + (nc * H + 2 * y2) * W + 2 * x2;
  out[i] = fmaxf(fmaxf(p[0], p[1]), fmaxf(p[W], p[W + 1]));
}

__global__ void k_convt2_nchw(const float* __restrict__ in, const float* __restrict__ w,
                              const float* __restrict__ bias, float* __restrict__ out,
                              int Nn, int Ci, int h, int ww, int Co, int coff, int ctot) {
  int H2 = 2 * h, W2 = 2 * ww;
  long long n = (long long)Nn * Co * H2 * W2, i = GTID; if (i >= n) return;
  int X = i % W2; long long t = i / W2; int Y = t % H2; t /= H2; int o = t % Co; int nb = (int)(t / Co);
  int y = Y >> 1, dy = Y & 1, x = X >> 1, dx = X & 1;
  float s = bias[o];
  for (int ci = 0; ci < Ci; ++ci)
    s += in[(((long long)nb * Ci + ci) * h + y) * ww + x]
       * w[(((long long)ci * Co + o) * 2 + dy) * 2 + dx];
  out[(((long long)nb * ctot + coff + o) * H2 + Y) * W2 + X] = s;
}

__global__ void k_copy_nchw_ch(const float* __restrict__ src, float* __restrict__ dst,
                               int Nn, int Cs, int H, int W, int coff, int ctot) {
  long long n = (long long)Nn * Cs * H * W, i = GTID; if (i >= n) return;
  int x = i % W; long long t = i / W; int y = t % H; t /= H; int c = t % Cs; int nb = (int)(t / Cs);
  dst[(((long long)nb * ctot + coff + c) * H + y) * W + x] = src[i];
}

__global__ void k_paste(const float* __restrict__ coarse, const float* __restrict__ ref,
                        const float* __restrict__ maxv, const int* __restrict__ maxi,
                        float* __restrict__ out, int H, int W, long long n) {
  long long i = GTID; if (i >= n) return;
  int x = i % W; long long t = i / W; int y = t % H; int bk = (int)(t / H);
  float v = coarse[i];
  int idx = maxi[bk]; int cy = idx / W, cx = idx % W;
  int top = imax(cy - 16, 0), bottom = imin(cy + 16, H);
  int left = imax(cx - 16, 0), right = imin(cx + 16, W);
  if (y >= top && y < bottom && x >= left && x < right && maxv[bk] >= 0.3f) {
    float Ly = (float)(bottom - top), Lx = (float)(right - left);
    float sy = fminf(fmaxf((y - top + 0.5f) * 32.0f / Ly - 0.5f, 0.0f), 31.0f);
    float sx = fminf(fmaxf((x - left + 0.5f) * 32.0f / Lx - 0.5f, 0.0f), 31.0f);
    int y0 = (int)floorf(sy), x0 = (int)floorf(sx);
    int y1 = imin(y0 + 1, 31), x1 = imin(x0 + 1, 31);
    float wy = sy - y0, wx = sx - x0;
    const float* p = ref + (long long)bk * 1024;
    v = (1 - wy) * ((1 - wx) * p[y0 * 32 + x0] + wx * p[y0 * 32 + x1])
      +      wy  * ((1 - wx) * p[y1 * 32 + x0] + wx * p[y1 * 32 + x1]);
  }
  out[i] = v;
}

// ---------------------------------------------------------------------------
// Host orchestration
// ---------------------------------------------------------------------------

extern "C" void kernel_launch(void* const* d_in, const int* in_sizes, int n_in,
                              void* d_out, int out_size, void* d_ws, size_t ws_size,
                              hipStream_t stream) {
  (void)out_size; (void)ws_size;
  if (n_in < 127) return;
  const float* IN[127];
  for (int i = 0; i < 127; ++i) IN[i] = (const float*)d_in[i];

  const int B = 16;
  char* wp = (char*)d_ws; size_t off = 0;
  auto alloc = [&](size_t bytes) -> void* {
    void* p = wp + off; off += (bytes + 255) & ~(size_t)255; return p;
  };
  auto af = [&](size_t n) { return (float*)alloc(n * 4); };
  auto ah = [&](size_t n) { return (half_t*)alloc(n * 2); };
  auto ai = [&](size_t n) { return (int*)alloc(n * 4); };

  float* xn    = af(3145728);        // x NHWC
  float* dwout = af(50331648);       // depthwise output arena (max dec1)
  float* e1 = af(25165824); float* e2 = af(12582912);
  float* e3 = af(4718592);  float* e4 = af(2359296);
  float* poolb = af(6291456);
  float* gnbuf = af(256); float* cmean = af(16 * 288); float* cscale = af(16 * 288);
  half_t* fA = ah(70254592);         // K-padded A arena (max: 1M rows x Kp=64 + slack)
  half_t* fW = ah(1048576);
  float* tmp = af(25165824);
  float* ybuf = af(589824); float* hbuf = af(589824); float* trows = af(589824);
  float* qkvb = af(1769472); float* obuf = af(589824); float* hh = af(1179648);
  float* cat288 = af(1179648); float* mv1o = af(589824); float* mv2o = af(589824);
  float* cat = af(50331648);
  float* d4 = af(2359296); float* d3 = af(4718592); float* d2 = af(12582912); float* d1 = af(25165824);
  float* cnhwc = af(17825792); float* coarse = af(17825792);
  float* maxv = af(272); int* maxi = ai(272);
  float* crop = af(278528); float* u1 = af(4456448); float* upl = af(1114112);
  float* u2 = af(2228224); float* ucat = af(8912896); float* ud = af(4456448);
  float* uref = af(278528);

  auto blocks = [](long long n) { return dim3((unsigned)((n + 255) / 256)); };
  auto kpad = [](int K) { return (K + 31) & ~31; };
  auto castPad = [&](const float* s, half_t* d, long long rows, int K) {
    int Kp = kpad(K);
    k_cast_pad<<<blocks(rows * (long long)Kp), 256, 0, stream>>>(s, d, rows, K, Kp);
  };
  auto gemm = [&](const half_t* A, const half_t* W_, const float* bias, float* C,
                  long long M, int N, int K, int ldc, int act) {
    int Kp = kpad(K);
    dim3 g((unsigned)((M + 63) / 64), (unsigned)((N + 63) / 64));
    k_gemm_wmma<<<g, 256, 0, stream>>>(A, W_, bias, C, (int)M, N, Kp, ldc, act);
  };

  // dsconv: dw 3x3 -> 1x1 WMMA GEMM -> GN+ReLU -> ECA
  auto dsconv = [&](const float* in, int H, int W, int Ci, int Co,
                    const float* dw, const float* pw, const float* gw, const float* gb,
                    const float* ew, int ek, float* out) {
    long long npix = (long long)B * H * W;
    k_dwconv3x3<<<blocks(npix * Ci), 256, 0, stream>>>(in, dw, dwout, B, H, W, Ci);
    castPad(dwout, fA, npix, Ci);
    castPad(pw, fW, Co, Ci);
    gemm(fA, fW, nullptr, out, npix, Co, Ci, Co, 0);
    int cg = Co / 8;
    k_gn_stats<<<dim3(B * 8), 256, 0, stream>>>(out, gnbuf, H * W, Co, cg);
    k_gn_apply_relu<<<blocks(npix * Co), 256, 0, stream>>>(out, gnbuf, gw, gb, out,
                                                           npix * Co, H * W, Co, cg);
    k_chan_mean<<<dim3(B * Co), 256, 0, stream>>>(out, cmean, H * W, Co);
    k_eca_scale<<<blocks((long long)B * Co), 256, 0, stream>>>(cmean, ew, cscale, B, Co, ek);
    k_scale_ch<<<blocks(npix * Co), 256, 0, stream>>>(out, cscale, npix * Co, H * W, Co);
  };

  auto tblock = [&](float* t, int pb) {           // one transformer layer on trows
    const long long R = 4096; const int D = 144;
    k_layernorm<<<dim3((unsigned)R), 64, 0, stream>>>(t, IN[pb + 0], IN[pb + 1], hbuf, D);
    castPad(hbuf, fA, R, D); castPad(IN[pb + 2], fW, 432, D);
    gemm(fA, fW, IN[pb + 3], qkvb, R, 432, D, 432, 0);
    k_attn<<<dim3(64, 4, 64), 64, 0, stream>>>(qkvb, obuf);
    castPad(obuf, fA, R, D); castPad(IN[pb + 4], fW, D, D);
    gemm(fA, fW, IN[pb + 5], tmp, R, D, D, D, 0);
    k_add_inplace<<<blocks(R * D), 256, 0, stream>>>(t, tmp, R * D);
    k_layernorm<<<dim3((unsigned)R), 64, 0, stream>>>(t, IN[pb + 6], IN[pb + 7], hbuf, D);
    castPad(hbuf, fA, R, D); castPad(IN[pb + 8], fW, 288, D);
    gemm(fA, fW, IN[pb + 9], hh, R, 288, D, 288, 2);
    castPad(hh, fA, R, 288); castPad(IN[pb + 10], fW, D, 288);
    gemm(fA, fW, IN[pb + 11], tmp, R, D, 288, D, 0);
    k_add_inplace<<<blocks(R * D), 256, 0, stream>>>(t, tmp, R * D);
  };

  auto mobilevit = [&](const float* in, int pb, float* out) {
    const long long M = 4096; const int C = 144;
    int Kp9 = kpad(9 * C);
    k_im2col3<<<blocks(M * Kp9), 256, 0, stream>>>(in, fA, B, 16, 16, C, Kp9);
    k_pack_w3<<<blocks((long long)C * Kp9), 256, 0, stream>>>(IN[pb + 0], fW, C, C, Kp9);
    gemm(fA, fW, IN[pb + 1], ybuf, M, C, 9 * C, C, 2);            // kw conv + silu
    castPad(ybuf, fA, M, C); castPad(IN[pb + 2], fW, C, C);
    gemm(fA, fW, nullptr, hbuf, M, C, C, C, 0);                   // pw1
    k_unfold<<<blocks(M * C), 256, 0, stream>>>(hbuf, trows, C);
    tblock(trows, pb + 3);
    tblock(trows, pb + 15);
    k_layernorm<<<dim3((unsigned)M), 64, 0, stream>>>(trows, IN[pb + 27], IN[pb + 28], hbuf, C);
    k_fold<<<blocks(M * C), 256, 0, stream>>>(hbuf, ybuf, C);
    castPad(ybuf, fA, M, C); castPad(IN[pb + 29], fW, C, C);
    gemm(fA, fW, IN[pb + 30], hbuf, M, C, C, C, 2);               // proj + silu
    k_copy_ch<<<blocks(M * C), 256, 0, stream>>>(in, cat288, M, C, 0, 2 * C);
    k_copy_ch<<<blocks(M * C), 256, 0, stream>>>(hbuf, cat288, M, C, C, 2 * C);
    int Kp18 = kpad(18 * C);
    k_im2col3<<<blocks(M * Kp18), 256, 0, stream>>>(cat288, fA, B, 16, 16, 2 * C, Kp18);
    k_pack_w3<<<blocks((long long)C * Kp18), 256, 0, stream>>>(IN[pb + 31], fW, C, 2 * C, Kp18);
    gemm(fA, fW, IN[pb + 32], out, M, C, 18 * C, C, 2);           // fuse conv + silu
  };

  auto up = [&](const float* in, int h, int w, int Ci, int Co,
                const float* uw, const float* ub, const float* skip, float* catb) {
    long long Mi = (long long)B * h * w;
    castPad(in, fA, Mi, Ci);
    int Kp = kpad(Ci);
    k_pack_wt2<<<blocks(4LL * Co * Kp), 256, 0, stream>>>(uw, fW, Ci, Co, Kp);
    gemm(fA, fW, nullptr, tmp, Mi, 4 * Co, Ci, 4 * Co, 0);
    k_convt_scatter<<<blocks(Mi * 4 * Co), 256, 0, stream>>>(tmp, ub, catb, B, h, w, Co, 0, 2 * Co);
    k_copy_ch<<<blocks((long long)B * 4 * h * w * Co), 256, 0, stream>>>(
        skip, catb, (long long)B * 4 * h * w, Co, Co, 2 * Co);
  };

  // ----- encoder -----
  k_nchw_to_nhwc<<<blocks(3145728), 256, 0, stream>>>(IN[0], xn, B, 3, 256, 256);
  dsconv(xn, 256, 256, 3, 24, IN[1], IN[2], IN[3], IN[4], IN[5], in_sizes[5], e1);
  k_pool2_nhwc<<<blocks(6291456), 256, 0, stream>>>(e1, poolb, B, 256, 256, 24);
  dsconv(poolb, 128, 128, 24, 48, IN[6], IN[7], IN[8], IN[9], IN[10], in_sizes[10], e2);
  k_pool2_nhwc<<<blocks(3145728), 256, 0, stream>>>(e2, poolb, B, 128, 128, 48);
  dsconv(poolb, 64, 64, 48, 72, IN[11], IN[12], IN[13], IN[14], IN[15], in_sizes[15], e3);
  k_pool2_nhwc<<<blocks(1179648), 256, 0, stream>>>(e3, poolb, B, 64, 64, 72);
  dsconv(poolb, 32, 32, 72, 144, IN[16], IN[17], IN[18], IN[19], IN[20], in_sizes[20], e4);
  k_pool2_nhwc<<<blocks(589824), 256, 0, stream>>>(e4, poolb, B, 32, 32, 144);

  // ----- bottleneck -----
  mobilevit(poolb, 21, mv1o);
  mobilevit(mv1o, 54, mv2o);

  // ----- decoder -----
  up(mv2o, 16, 16, 144, 144, IN[87], IN[88], e4, cat);
  dsconv(cat, 32, 32, 288, 144, IN[89], IN[90], IN[91], IN[92], IN[93], in_sizes[93], d4);
  up(d4, 32, 32, 144, 72, IN[94], IN[95], e3, cat);
  dsconv(cat, 64, 64, 144, 72, IN[96], IN[97], IN[98], IN[99], IN[100], in_sizes[100], d3);
  up(d3, 64, 64, 72, 48, IN[101], IN[102], e2, cat);
  dsconv(cat, 128, 128, 96, 48, IN[103], IN[104], IN[105], IN[106], IN[107], in_sizes[107], d2);
  up(d2, 128, 128, 48, 24, IN[108], IN[109], e1, cat);
  dsconv(cat, 256, 256, 48, 24, IN[110], IN[111], IN[112], IN[113], IN[114], in_sizes[114], d1);

  // ----- final 1x1 conv -> coarse -----
  long long Mf = (long long)B * 65536;
  castPad(d1, fA, Mf, 24);
  castPad(IN[115], fW, 17, 24);
  gemm(fA, fW, IN[116], cnhwc, Mf, 17, 24, 17, 0);
  k_nhwc_to_nchw<<<blocks(Mf * 17), 256, 0, stream>>>(cnhwc, coarse, B, 17, 256, 256);

  // ----- refine -----
  const int BK = B * 17;
  k_argmax<<<dim3(BK), 256, 0, stream>>>(coarse, maxv, maxi, 65536);
  k_crop<<<blocks((long long)BK * 1024), 256, 0, stream>>>(coarse, maxi, crop, 256, 256,
                                                           (long long)BK * 1024);
  k_conv_nchw<<<blocks((long long)BK * 16 * 1024), 256, 0, stream>>>(
      crop, IN[117], IN[118], u1, BK, 1, 32, 32, 16, 3, 1, 1);
  k_pool2_nchw<<<blocks((long long)BK * 16 * 256), 256, 0, stream>>>(u1, upl, BK, 16, 32, 32);
  k_conv_nchw<<<blocks((long long)BK * 32 * 256), 256, 0, stream>>>(
      upl, IN[119], IN[120], u2, BK, 16, 16, 16, 32, 3, 1, 1);
  k_convt2_nchw<<<blocks((long long)BK * 16 * 1024), 256, 0, stream>>>(
      u2, IN[121], IN[122], ucat, BK, 32, 16, 16, 16, 0, 32);
  k_copy_nchw_ch<<<blocks((long long)BK * 16 * 1024), 256, 0, stream>>>(
      u1, ucat, BK, 16, 32, 32, 16, 32);
  k_conv_nchw<<<blocks((long long)BK * 16 * 1024), 256, 0, stream>>>(
      ucat, IN[123], IN[124], ud, BK, 32, 32, 32, 16, 3, 1, 1);
  k_conv_nchw<<<blocks((long long)BK * 1024), 256, 0, stream>>>(
      ud, IN[125], IN[126], uref, BK, 16, 32, 32, 1, 1, 0, 0);
  k_paste<<<blocks((long long)BK * 65536), 256, 0, stream>>>(
      coarse, uref, maxv, maxi, (float*)d_out, 256, 256, (long long)BK * 65536);
}
